// Rank_IGR_Centerness_Loss_13967233646997
// MI455X (gfx1250) — matrix operand order repaired and checked by reading, not compiled, not run on hardware
//
#include <hip/hip_runtime.h>
#include <hip/hip_bf16.h>

typedef __attribute__((ext_vector_type(16))) __bf16 v16bf;
typedef __attribute__((ext_vector_type(8)))  float  v8f;

#define G1f 3.0f
#define G2f 5.0f
#define THRf 0.05f

// ---------------- Phase 1: per-element prep --------------------------------
// cI[n] = pos ? c : -1e30   (i-side sentinel: never selected)
// cJ[n] = pos ? c : +1e30   (j-side sentinel: c_i - cJ > THR never true)
// u = exp(-G1*p), v = exp(+G1*p) split into bf16 hi/lo
// w = exp(-G2*iou), x = exp(+G2*iou) split into bf16 hi/lo
__global__ void rankloss_prep(const float* __restrict__ cls,
                              const float* __restrict__ label_cls,
                              const float* __restrict__ label_loc,
                              const float* __restrict__ pred_bboxes,
                              const float* __restrict__ label_target,
                              const int*   __restrict__ flag_p,
                              float* __restrict__ cI, float* __restrict__ cJ,
                              float* __restrict__ uA, float* __restrict__ wA,
                              __bf16* __restrict__ vhi, __bf16* __restrict__ vlo,
                              __bf16* __restrict__ xhi, __bf16* __restrict__ xlo,
                              float* __restrict__ accE1, float* __restrict__ accE2,
                              float* __restrict__ accCnt,
                              int B, int N)
{
    int bn = blockIdx.x * blockDim.x + threadIdx.x;
    int BN = B * N;
    if (bn >= BN) return;
    if (bn < B) { accE1[bn] = 0.f; accE2[bn] = 0.f; accCnt[bn] = 0.f; }

    int b = bn / N;
    int n = bn - b * N;

    // centerness from label_loc (B,4,N): components 0,2 = lr ; 1,3 = tb
    const float* loc = label_loc + (size_t)b * 4 * N;
    float L  = loc[n];
    float T  = loc[N + n];
    float R  = loc[2 * N + n];
    float Bo = loc[3 * N + n];
    float cen = sqrtf((fminf(L, R) / fmaxf(L, R)) * (fminf(T, Bo) / fmaxf(T, Bo)));

    // prob from cls (B,N,2)
    float c1 = cls[(size_t)bn * 2 + 1];
    float prob = (*flag_p != 0) ? c1 : expf(c1);

    // IoU vs target
    const float* pb = pred_bboxes + (size_t)b * 4 * N;
    float x1 = pb[n], y1 = pb[N + n], x2 = pb[2 * N + n], y2 = pb[3 * N + n];
    const float* lt = label_target + (size_t)b * 4;
    float tx1 = lt[0], ty1 = lt[1], tx2 = lt[2], ty2 = lt[3];
    float ww = fmaxf(fminf(tx2, x2) - fmaxf(tx1, x1), 0.f);
    float hh = fmaxf(fminf(ty2, y2) - fmaxf(ty1, y1), 0.f);
    float area  = (x2 - x1) * (y2 - y1);
    float tarea = (tx2 - tx1) * (ty2 - ty1);
    float inter = ww * hh;
    float iou   = inter / (area + tarea - inter);

    bool pos = label_cls[bn] > 0.f;
    cI[bn] = pos ? cen : -1e30f;
    cJ[bn] = pos ? cen :  1e30f;

    float u = expf(-G1f * prob);
    float v = expf( G1f * prob);
    float w = expf(-G2f * iou);
    float x = expf( G2f * iou);
    uA[bn] = u;
    wA[bn] = w;

    __bf16 vh = (__bf16)v;  vhi[bn] = vh;  vlo[bn] = (__bf16)(v - (float)vh);
    __bf16 xh = (__bf16)x;  xhi[bn] = xh;  xlo[bn] = (__bf16)(x - (float)xh);
}

// ---------------- Phase 2: masked bilinear forms via WMMA ------------------
// One wave (32 threads) per 16-row I tile. Loops over j in chunks of 32.
// A (16x32 bf16) = mask tile, exactly 0/1.
// B (32x16 bf16) columns: 0=vhi 1=vlo 2=xhi 3=xlo 4=ones, rest 0.
// D accumulates row-sums in f32; epilogue dots rows with u / w and atomically
// accumulates per-batch e1, e2, cnt.
__global__ void rankloss_pairs(const float* __restrict__ cI,
                               const float* __restrict__ cJ,
                               const float* __restrict__ uA,
                               const float* __restrict__ wA,
                               const __bf16* __restrict__ vhi,
                               const __bf16* __restrict__ vlo,
                               const __bf16* __restrict__ xhi,
                               const __bf16* __restrict__ xlo,
                               float* __restrict__ accE1, float* __restrict__ accE2,
                               float* __restrict__ accCnt,
                               int N)
{
    const int tilesPerB = N >> 4;
    const int b    = blockIdx.x / tilesPerB;
    const int it   = blockIdx.x - b * tilesPerB;
    const int lane = threadIdx.x;          // wave32
    const int half = lane >> 4;            // 0 or 1
    const int nn   = lane & 15;            // A: row m ; B/D: column n
    const int baseI = b * N + it * 16;
    const int baseB = b * N;

    const float ci = cI[baseI + nn];       // this lane's row centerness (i-side)

    // B-column data source for this lane
    const __bf16* colp = vhi;
    if      (nn == 1) colp = vlo;
    else if (nn == 2) colp = xhi;
    else if (nn == 3) colp = xlo;
    const __bf16 one  = (__bf16)1.0f;
    const __bf16 zero = (__bf16)0.0f;
    const __bf16 fixedv  = (nn == 4) ? one : zero;
    const bool   useFixed = (nn >= 4);

    v8f acc = {0.f, 0.f, 0.f, 0.f, 0.f, 0.f, 0.f, 0.f};

    for (int jc = 0; jc < N; jc += 32) {
        // ---- A: 16x32 bf16 mask tile.
        // vector elem i -> K = i + 16*half
        // => one contiguous 16-run of cJ
        v16bf a;
        const float* cjp0 = cJ + baseB + jc + half * 16;
        const float* cjp1 = cjp0 + 8;
#pragma unroll
        for (int t = 0; t < 8; ++t) {
            a[t]     = (ci - cjp0[t] > THRf) ? one : zero;
            a[8 + t] = (ci - cjp1[t] > THRf) ? one : zero;
        }

        // ---- B: 32x16 bf16. Lane's column nn; vector elem s -> K = 16*half + s
        v16bf bv;
        if (!useFixed) {
            const __bf16* cp = colp + baseB + jc + half * 16;
#pragma unroll
            for (int s = 0; s < 16; ++s) bv[s] = cp[s];
        } else {
#pragma unroll
            for (int s = 0; s < 16; ++s) bv[s] = fixedv;
        }

        // D += A x B   (f32 accumulate)
        acc = __builtin_amdgcn_wmma_f32_16x16x32_bf16(
            /*neg_a=*/false, a, /*neg_b=*/false, bv,
            /*c_mod=*/(short)0, acc, /*reuse_a=*/false, /*reuse_b=*/false);
    }

    // ---- Epilogue: D vgpr r holds D[m][n], m = r + 8*half, n = lane&15.
    float s = 0.f;
    if (nn < 2) {                              // columns 0,1 -> e1 = sum u_m * (Rhi+Rlo)
#pragma unroll
        for (int r = 0; r < 8; ++r) s += uA[baseI + half * 8 + r] * acc[r];
        atomicAdd(&accE1[b], s);
    } else if (nn < 4) {                       // columns 2,3 -> e2
#pragma unroll
        for (int r = 0; r < 8; ++r) s += wA[baseI + half * 8 + r] * acc[r];
        atomicAdd(&accE2[b], s);
    } else if (nn == 4) {                      // column 4 -> exact pair count
#pragma unroll
        for (int r = 0; r < 8; ++r) s += acc[r];
        atomicAdd(&accCnt[b], s);
    }
}

// ---------------- Phase 3: finalize ---------------------------------------
__global__ void rankloss_final(const float* __restrict__ accE1,
                               const float* __restrict__ accE2,
                               const float* __restrict__ accCnt,
                               int B, float* __restrict__ out)
{
    if (threadIdx.x == 0 && blockIdx.x == 0) {
        float s1 = 0.f, s2 = 0.f, nv = 0.f;
        for (int b = 0; b < B; ++b) {
            float cnt = accCnt[b];
            if (cnt > 0.f) {
                float safe = fmaxf(cnt, 1.f);
                s1 += accE1[b] / safe;
                s2 += accE2[b] / safe;
                nv += 1.f;
            }
        }
        float d = fmaxf(nv, 1.f);
        out[0] = (nv > 0.f) ? s1 / d : 0.f;
        out[1] = (nv > 0.f) ? s2 / d : 0.f;
    }
}

// ---------------- Launcher -------------------------------------------------
extern "C" void kernel_launch(void* const* d_in, const int* in_sizes, int n_in,
                              void* d_out, int out_size, void* d_ws, size_t ws_size,
                              hipStream_t stream)
{
    const float* cls          = (const float*)d_in[0];
    const float* label_cls    = (const float*)d_in[1];
    const float* label_loc    = (const float*)d_in[2];
    const float* pred_bboxes  = (const float*)d_in[3];
    const float* label_target = (const float*)d_in[4];
    const int*   flag         = (const int*)d_in[5];

    const int B  = in_sizes[4] / 4;     // label_target is (B,4)
    const int N  = in_sizes[1] / B;     // label_cls is (B,N); N % 32 == 0
    const int BN = B * N;

    // workspace carve-out: 24*BN + 12*B bytes total (~400 KB for B=8,N=2048)
    float*  cI   = (float*)d_ws;
    float*  cJ   = cI + BN;
    float*  uA   = cJ + BN;
    float*  wA   = uA + BN;
    __bf16* vhi  = (__bf16*)(wA + BN);
    __bf16* vlo  = vhi + BN;
    __bf16* xhi  = vlo + BN;
    __bf16* xlo  = xhi + BN;
    float*  accE1  = (float*)(xlo + BN);
    float*  accE2  = accE1 + B;
    float*  accCnt = accE2 + B;

    rankloss_prep<<<(BN + 255) / 256, 256, 0, stream>>>(
        cls, label_cls, label_loc, pred_bboxes, label_target, flag,
        cI, cJ, uA, wA, vhi, vlo, xhi, xlo, accE1, accE2, accCnt, B, N);

    rankloss_pairs<<<B * (N / 16), 32, 0, stream>>>(
        cI, cJ, uA, wA, vhi, vlo, xhi, xlo, accE1, accE2, accCnt, N);

    rankloss_final<<<1, 1, 0, stream>>>(accE1, accE2, accCnt, B, (float*)d_out);
}
